// PartStructureLayer_52561809769061
// MI455X (gfx1250) — matrix operand order repaired and checked by reading, not compiled, not run on hardware
//
#include <hip/hip_runtime.h>
#include <hip/hip_bf16.h>

// ---------------------------------------------------------------------------
// MI455X / gfx1250 implementation.
// Heavy GEMMs + attention run on v_wmma_f32_16x16x32_bf16 (f32 accumulate).
// GEMM tiles are staged into LDS with GLOBAL_LOAD_ASYNC_TO_LDS_B128 (the
// CDNA5 direct-to-LDS async path, ASYNCcnt-tracked) instead of going through
// VGPRs. The "part structure" branch is algebraically collapsed (pw = m m^T
// is rank-1 and m >= 0), leaving only reductions + one 512x768 GEMV.
// ---------------------------------------------------------------------------

typedef __attribute__((ext_vector_type(16))) __bf16 v16bf;
typedef __attribute__((ext_vector_type(8)))  float  v8f;

#define WMMA_BF16(a, b, c) \
  __builtin_amdgcn_wmma_f32_16x16x32_bf16(false, (a), false, (b), (short)0, (c), false, false)

#define BATCH 16
#define SEQ   577
#define CDIM  768
#define NHEAD 12
#define HDIM  64
#define MROWS (BATCH * SEQ)   // 9232
#define NKPAD 608             // 19 * 32, padded key dim for PV wmma
#define SSTR  612             // LDS stride (floats) for scores, bank-skewed
#define PSTR  616             // LDS stride (bf16) for probs, bank-skewed

// float -> bf16, round-to-nearest-even (storage-only __bf16, no bf16 ALU)
__device__ __forceinline__ __bf16 f2bf(float f) {
  union { float f; unsigned u; } x; x.f = f;
  unsigned r = x.u + 0x7FFFu + ((x.u >> 16) & 1u);
  unsigned short h = (unsigned short)(r >> 16);
  return __builtin_bit_cast(__bf16, h);
}

// Build a 32-byte WMMA fragment from two 16-byte chunks (global or LDS).
__device__ __forceinline__ v16bf ld_frag(const __bf16* p0, const __bf16* p1) {
  v16bf r;
  ((uint4*)&r)[0] = *(const uint4*)p0;
  ((uint4*)&r)[1] = *(const uint4*)p1;
  return r;
}

// CDNA5 async direct-to-LDS 128-bit load (GV mode, tracked by ASYNCcnt).
// ldsoff = LDS byte address (low 32 bits of the generic shared-aperture
// address), gaddr = 64-bit global byte address. Data never touches VGPRs.
__device__ __forceinline__ void async_b128(unsigned ldsoff, const void* gaddr) {
  asm volatile("global_load_async_to_lds_b128 %0, %1, off"
               :: "v"(ldsoff), "v"((unsigned long long)(uintptr_t)gaddr)
               : "memory");
}
__device__ __forceinline__ void wait_async0() {
  asm volatile("s_wait_asynccnt 0x0" ::: "memory");
}

// ---------------------------------------------------------------------------
// Elementwise f32 -> bf16
// ---------------------------------------------------------------------------
__global__ void k_cvt(const float* __restrict__ src, __bf16* __restrict__ dst, int n) {
  for (int i = blockIdx.x * blockDim.x + threadIdx.x; i < n; i += gridDim.x * blockDim.x)
    dst[i] = f2bf(src[i]);
}

// src: rows x cols (row-major f32) -> dst: cols x rows (row-major bf16) = src^T
__global__ void k_transpose(const float* __restrict__ src, __bf16* __restrict__ dst,
                            int rows, int cols) {
  int n = rows * cols;
  for (int i = blockIdx.x * blockDim.x + threadIdx.x; i < n; i += gridDim.x * blockDim.x) {
    int c = i / rows, r = i - c * rows;
    dst[i] = f2bf(src[r * cols + c]);
  }
}

// ---------------------------------------------------------------------------
// Tiled bf16 WMMA GEMM: C[M x N] = A[M x 768] * BT[N x 768]^T + bias
//   MODE 0: QKV epilogue -> scatter into q (B,H,N,64), k (B,H,N,64),
//           vT (B,H,64,NKPAD), all bf16
//   MODE 1: proj epilogue -> f32 output (d_out)
// Block: 128 threads = 4 waves; block tile 128x64; each wave owns a 32-row
// strip (2 A fragments reused across 4 B fragments -> 8 wmma per K-step).
// Tiles filled with async direct-to-LDS B128 loads; stride 48 halves keeps
// fragments 16B-aligned and skews banks. Guarded tail rows simply do not
// issue loads: stale LDS only feeds C rows that the epilogue discards.
// ---------------------------------------------------------------------------
template <int MODE>
__global__ __launch_bounds__(128) void k_gemm(const __bf16* __restrict__ A,
                                              const __bf16* __restrict__ BT,
                                              const float* __restrict__ bias,
                                              __bf16* __restrict__ qo,
                                              __bf16* __restrict__ ko,
                                              __bf16* __restrict__ vT,
                                              float* __restrict__ outf) {
  __shared__ __bf16 As[128 * 48];
  __shared__ __bf16 Bs[64 * 48];

  const int tid = threadIdx.x, wave = tid >> 5, lane = tid & 31;
  const int m0 = blockIdx.x * 128, n0 = blockIdx.y * 64;

  const v8f z8 = {0.f, 0.f, 0.f, 0.f, 0.f, 0.f, 0.f, 0.f};
  v8f acc[2][4] = {{z8, z8, z8, z8}, {z8, z8, z8, z8}};

  const int arow0 = wave * 32 + (lane & 15);
  const int akb  = (lane < 16) ? 0 : 8;    // A frag K split {0..7,16..23}/{8..15,24..31}
  const int bcol = (lane & 15);
  const int bkb  = (lane < 16) ? 0 : 8;    // B frag K split {0..7,16..23}/{8..15,24..31}

  for (int k0 = 0; k0 < 768; k0 += 32) {
    // A tile: 128 rows x 4 x 16B  (async, EXEC-masked on the M tail)
    for (int t = tid; t < 512; t += 128) {
      int row = t >> 2, quad = t & 3;
      if (m0 + row < MROWS)
        async_b128((unsigned)(uintptr_t)&As[row * 48 + quad * 8],
                   A + (size_t)(m0 + row) * 768 + k0 + quad * 8);
    }
    // B tile: 64 rows x 4 x 16B (always in range)
    for (int t = tid; t < 256; t += 128) {
      int row = t >> 2, quad = t & 3;
      async_b128((unsigned)(uintptr_t)&Bs[row * 48 + quad * 8],
                 BT + (size_t)(n0 + row) * 768 + k0 + quad * 8);
    }
    wait_async0();
    __syncthreads();

    v16bf a0 = ld_frag(&As[arow0 * 48 + akb],        &As[arow0 * 48 + akb + 16]);
    v16bf a1 = ld_frag(&As[(arow0 + 16) * 48 + akb], &As[(arow0 + 16) * 48 + akb + 16]);
#pragma unroll
    for (int j = 0; j < 4; j++) {
      const __bf16* bp = &Bs[(j * 16 + bcol) * 48 + bkb];
      v16bf bf_ = ld_frag(bp, bp + 16);
      acc[0][j] = WMMA_BF16(a0, bf_, acc[0][j]);
      acc[1][j] = WMMA_BF16(a1, bf_, acc[1][j]);
    }
    __syncthreads();
  }

  const int mo = (lane >= 16) ? 8 : 0;      // C layout: M = r + 8*(lane>=16)
#pragma unroll
  for (int fr = 0; fr < 2; fr++) {
#pragma unroll
    for (int j = 0; j < 4; j++) {
#pragma unroll
      for (int r = 0; r < 8; r++) {
        int gr = m0 + wave * 32 + fr * 16 + mo + r;
        if (gr >= MROWS) continue;
        int gc = n0 + j * 16 + (lane & 15);
        float v = acc[fr][j][r] + bias[gc];
        if (MODE == 0) {
          int three = gc / 768;
          int rem = gc - three * 768;
          int h = rem >> 6, dd = rem & 63;
          int b = gr / 577, n = gr - b * 577;
          size_t bh = (size_t)(b * 12 + h);
          if (three == 0)      qo[(bh * 577 + n) * 64 + dd] = f2bf(v);
          else if (three == 1) ko[(bh * 577 + n) * 64 + dd] = f2bf(v);
          else                 vT[(bh * 64 + dd) * NKPAD + n] = f2bf(v);  // v transposed
        } else {
          outf[(size_t)gr * 768 + gc] = v;
        }
      }
    }
  }
}

// ---------------------------------------------------------------------------
// Attention: one block per (16-query chunk, head, batch). 256 threads = 8
// waves. Phase 1: 8 waves tile QK^T across 37 key tiles (2 wmma each).
// Phase 2: block softmax (full 577-row in LDS), probs -> bf16 in LDS,
// query-0 row emits amap. Phase 3: 4 waves do P*V (19 wmma each), store
// context bf16 (B,N,768).
// ---------------------------------------------------------------------------
__global__ __launch_bounds__(256) void k_attn(const __bf16* __restrict__ q,
                                              const __bf16* __restrict__ k,
                                              const __bf16* __restrict__ vT,
                                              __bf16* __restrict__ ctx,
                                              float* __restrict__ amap) {
  __shared__ float  sc[16 * SSTR];
  __shared__ __bf16 pr[16 * PSTR];
  __shared__ float  red[16 * 16];
  __shared__ float  rowmax[16];
  __shared__ float  rowsum[16];

  const int tid = threadIdx.x, wave = tid >> 5, lane = tid & 31;
  const int qc = blockIdx.x, h = blockIdx.y, b = blockIdx.z;
  const int q0 = qc * 16;
  const size_t bh = (size_t)(b * 12 + h);
  const float scale = 0.125f;               // 64^-0.5 (applied to scores)

  // ---- query fragments (K = 64 => two 16x32 fragments) ----
  int qrow = q0 + (lane & 15); if (qrow > 576) qrow = 576;
  const __bf16* qp = q + (bh * 577 + qrow) * 64;
  const int akb = (lane < 16) ? 0 : 8;
  v16bf a0 = ld_frag(qp + akb,      qp + akb + 16);
  v16bf a1 = ld_frag(qp + 32 + akb, qp + 32 + akb + 16);

  // ---- phase 1: scores = (q @ k^T) * scale ----
  const v8f z8 = {0.f, 0.f, 0.f, 0.f, 0.f, 0.f, 0.f, 0.f};
  for (int kt = wave; kt < 37; kt += 8) {
    int krow = kt * 16 + (lane & 15);
    int krc = (krow > 576) ? 576 : krow;
    const __bf16* kp = k + (bh * 577 + krc) * 64;
    int ks = (lane < 16) ? 0 : 8;
    v16bf b0 = ld_frag(kp + ks,      kp + ks + 16);
    v16bf b1 = ld_frag(kp + 32 + ks, kp + 32 + ks + 16);
    v8f s = z8;
    s = WMMA_BF16(a0, b0, s);
    s = WMMA_BF16(a1, b1, s);
    int mo = (lane >= 16) ? 8 : 0;
#pragma unroll
    for (int r = 0; r < 8; r++)
      sc[(mo + r) * SSTR + krow] = (krow < 577) ? s[r] * scale : -1e30f;
  }
  __syncthreads();

  // ---- phase 2: softmax over 577 keys, probs -> bf16, amap for query 0 ----
  const int row = tid >> 4, sub = tid & 15;
  float mx = -1e30f;
  for (int j = sub; j < 577; j += 16) mx = fmaxf(mx, sc[row * SSTR + j]);
  red[row * 16 + sub] = mx;
  __syncthreads();
  if (sub == 0) {
    float m2 = -1e30f;
    for (int i = 0; i < 16; i++) m2 = fmaxf(m2, red[row * 16 + i]);
    rowmax[row] = m2;
  }
  __syncthreads();
  const float rm = rowmax[row];
  float sum = 0.f;
  for (int j = sub; j < 577; j += 16) sum += __expf(sc[row * SSTR + j] - rm);
  red[row * 16 + sub] = sum;
  __syncthreads();
  if (sub == 0) {
    float s2 = 0.f;
    for (int i = 0; i < 16; i++) s2 += red[row * 16 + i];
    rowsum[row] = s2;
  }
  __syncthreads();
  const float inv = 1.0f / rowsum[row];
  const bool isrow0 = (q0 + row == 0);
  for (int j = sub; j < NKPAD; j += 16) {
    float p = (j < 577) ? __expf(sc[row * SSTR + j] - rm) * inv : 0.0f;
    pr[row * PSTR + j] = f2bf(p);
    if (isrow0 && j >= 1 && j < 577) amap[bh * 576 + (j - 1)] = p;  // attn[:,:,0,1:]
  }
  __syncthreads();

  // ---- phase 3: out = P @ V (waves 0..3, one 16-dim chunk each) ----
  if (wave < 4) {
    const int dim = wave * 16 + (lane & 15);
    const __bf16* vp = vT + (bh * 64 + dim) * NKPAD;
    const int pm = lane & 15;
    const int pkb = (lane < 16) ? 0 : 8;
    const int vks = (lane < 16) ? 0 : 8;
    v8f acc = z8;
    for (int kt = 0; kt < 19; kt++) {
      const __bf16* pp = &pr[pm * PSTR + kt * 32 + pkb];
      v16bf a = ld_frag(pp, pp + 16);
      v16bf bb = ld_frag(vp + kt * 32 + vks, vp + kt * 32 + vks + 16);
      acc = WMMA_BF16(a, bb, acc);
    }
    const int mo = (lane >= 16) ? 8 : 0;
#pragma unroll
    for (int r = 0; r < 8; r++) {
      int qr = q0 + mo + r;
      if (qr < 577)
        ctx[((size_t)b * 577 + qr) * 768 + h * 64 + dim] = f2bf(acc[r]);
    }
  }
}

// ---------------------------------------------------------------------------
// Structure branch, algebraically collapsed. One block per batch.
//   mask = sum_h amap; thr = mean; m = binary*mask/12 (m >= 0)
//   t_k = (sum_j m_j dist_j)*w_g1[0,k] + (sum_j m_j ang_j)*w_g1[1,k]
//   h1[j,k] = relu(lrelu(m_j t_k)) = m_j relu(t_k)   (m_j >= 0)
//   u_k = sum_j m_j h1[j,k] = (sum m_j^2) relu(t_k)
//   structure_c = lrelu( m[idx] * sum_k u_k w_g2[k,c] ); out[b,0,:] += structure
// ---------------------------------------------------------------------------
__global__ __launch_bounds__(256) void k_struct(const float* __restrict__ amap,
                                                const float* __restrict__ wg1,
                                                const float* __restrict__ wg2,
                                                float* __restrict__ out) {
  __shared__ float mask[576];
  __shared__ float mv[576];
  __shared__ float u[512];
  __shared__ float redv[256];
  __shared__ int   redi[256];

  const int b = blockIdx.x, tid = threadIdx.x;

  for (int j = tid; j < 576; j += 256) {
    float s = 0.f;
#pragma unroll
    for (int h = 0; h < 12; h++) s += amap[((size_t)b * 12 + h) * 576 + j];
    mask[j] = s;
  }
  __syncthreads();

  float part = 0.f;
  for (int j = tid; j < 576; j += 256) part += mask[j];
  redv[tid] = part; __syncthreads();
  for (int s = 128; s > 0; s >>= 1) { if (tid < s) redv[tid] += redv[tid + s]; __syncthreads(); }
  const float thr = redv[0] * (1.0f / 576.0f);
  __syncthreads();

  for (int j = tid; j < 576; j += 256) {
    float bin = (mask[j] > thr) ? 1.0f : 0.0f;
    mv[j] = bin * mask[j] * (1.0f / 12.0f);
  }
  __syncthreads();

  // argmax (first max wins, matching jnp.argmax)
  float bv = -1e30f; int bi = 0;
  for (int j = tid; j < 576; j += 256)
    if (mv[j] > bv) { bv = mv[j]; bi = j; }
  redv[tid] = bv; redi[tid] = bi; __syncthreads();
  for (int s = 128; s > 0; s >>= 1) {
    if (tid < s) {
      float ov = redv[tid + s]; int oi = redi[tid + s];
      if (ov > redv[tid] || (ov == redv[tid] && oi < redi[tid])) { redv[tid] = ov; redi[tid] = oi; }
    }
    __syncthreads();
  }
  const int idx = redi[0];
  const float mval = mv[idx];
  __syncthreads();

  const float ax = (float)(idx / 24), ay = (float)(idx % 24);
  float pa = 0.f, pb = 0.f, pc = 0.f;
  for (int j = tid; j < 576; j += 256) {
    float mj = mv[j];
    float bin = (mask[j] > thr) ? 1.0f : 0.0f;
    float rx = ((float)(j / 24) - ax) * (1.0f / 24.0f);
    float ry = ((float)(j % 24) - ay) * (1.0f / 24.0f);
    float dist = sqrtf(rx * rx + ry * ry) * bin;
    float ang  = (atan2f(ry, rx) * 0.15915494309189535f + 0.5f) * bin;
    pa += mj * dist; pb += mj * ang; pc += mj * mj;
  }
  redv[tid] = pa; __syncthreads();
  for (int s = 128; s > 0; s >>= 1) { if (tid < s) redv[tid] += redv[tid + s]; __syncthreads(); }
  const float Ad = redv[0]; __syncthreads();
  redv[tid] = pb; __syncthreads();
  for (int s = 128; s > 0; s >>= 1) { if (tid < s) redv[tid] += redv[tid + s]; __syncthreads(); }
  const float Bd = redv[0]; __syncthreads();
  redv[tid] = pc; __syncthreads();
  for (int s = 128; s > 0; s >>= 1) { if (tid < s) redv[tid] += redv[tid + s]; __syncthreads(); }
  const float c2 = redv[0]; __syncthreads();

  for (int kk = tid; kk < 512; kk += 256) {
    float tk = Ad * wg1[kk] + Bd * wg1[512 + kk];
    u[kk] = (tk > 0.f) ? c2 * tk : 0.f;
  }
  __syncthreads();

  for (int c = tid; c < 768; c += 256) {
    float z = 0.f;
    for (int kk = 0; kk < 512; kk++) z += u[kk] * wg2[kk * 768 + c];
    z *= mval;
    float st = (z > 0.f) ? z : 0.2f * z;   // lrelu
    out[(size_t)b * 577 * 768 + c] += st;  // token 0
  }
}

// ---------------------------------------------------------------------------
extern "C" void kernel_launch(void* const* d_in, const int* in_sizes, int n_in,
                              void* d_out, int out_size, void* d_ws, size_t ws_size,
                              hipStream_t stream) {
  const float* x      = (const float*)d_in[0];
  const float* w_qkv  = (const float*)d_in[1];
  const float* b_qkv  = (const float*)d_in[2];
  const float* w_proj = (const float*)d_in[3];
  const float* b_proj = (const float*)d_in[4];
  const float* w_g1   = (const float*)d_in[5];
  const float* w_g2   = (const float*)d_in[6];
  float* out = (float*)d_out;
  (void)in_sizes; (void)n_in; (void)out_size; (void)ws_size;

  char* p = (char*)d_ws;
  auto take = [&](size_t bytes) -> void* {
    void* r = (void*)p;
    p += (bytes + 255) & ~(size_t)255;
    return r;
  };
  __bf16* xb     = (__bf16*)take((size_t)MROWS * 768 * 2);          // also reused as ctx
  __bf16* wqkvT  = (__bf16*)take((size_t)2304 * 768 * 2);
  __bf16* wprojT = (__bf16*)take((size_t)768 * 768 * 2);
  __bf16* qb     = (__bf16*)take((size_t)16 * 12 * 577 * 64 * 2);
  __bf16* kb     = (__bf16*)take((size_t)16 * 12 * 577 * 64 * 2);
  __bf16* vT     = (__bf16*)take((size_t)16 * 12 * 64 * NKPAD * 2);
  float*  amap   = (float*)take((size_t)16 * 12 * 576 * 4);
  __bf16* ctx    = xb;  // xb is dead after the QKV GEMM; reuse for context

  k_cvt<<<2048, 256, 0, stream>>>(x, xb, MROWS * 768);
  k_transpose<<<2048, 256, 0, stream>>>(w_qkv, wqkvT, 768, 2304);
  k_transpose<<<1024, 256, 0, stream>>>(w_proj, wprojT, 768, 768);
  k_gemm<0><<<dim3(73, 36), 128, 0, stream>>>(xb, wqkvT, b_qkv, qb, kb, vT, nullptr);
  k_attn<<<dim3(37, 12, 16), 256, 0, stream>>>(qb, kb, vT, ctx, amap);
  k_gemm<1><<<dim3(73, 12), 128, 0, stream>>>(ctx, wprojT, b_proj, nullptr, nullptr, nullptr, out);
  k_struct<<<16, 256, 0, stream>>>(amap, w_g1, w_g2, out);
}
